// InfoNCEWithQueue_4595615006903
// MI455X (gfx1250) — matrix operand order repaired and checked by reading, not compiled, not run on hardware
//
#include <hip/hip_runtime.h>
#include <hip/hip_bf16.h>
#include <math.h>

// ---------------------------------------------------------------------------
// InfoNCE with queue, MI455X (gfx1250, wave32, WMMA bf16).
// query[256,768] f32, keys[256,768] f32, queue[64,1024,768] f32 -> scalar loss
//
// cos(q,c) = (q_raw . c_raw) * invq * invc  -> scales kept in f32,
// raw dot done in bf16 WMMA with f32 accumulate.
// Softmax is fused into the GEMM epilogue (per-row online (max,sum) partials),
// so the 256 x 65792 logit matrix is never materialized.
// ---------------------------------------------------------------------------

#define DD    768
#define BB    256
#define NNEG  65536
#define NCOLS (BB + NNEG)    // 65792
#define NBLK  (NCOLS / 64)   // 1028 column blocks
#define INV_TEMP 20.0f
#define EPS 1e-8f

typedef __attribute__((ext_vector_type(16))) __bf16 v16bf;
typedef __attribute__((ext_vector_type(8)))  __bf16 v8bf;
typedef __attribute__((ext_vector_type(8)))  float  v8f;
typedef __attribute__((ext_vector_type(4)))  float  v4f;

// ---------------------------------------------------------------------------
// Kernel 1: query -> bf16 + per-row inverse norm (f32).
// ---------------------------------------------------------------------------
__global__ __launch_bounds__(256) void prep_q_kernel(
    const float* __restrict__ q, __bf16* __restrict__ qbf,
    float* __restrict__ invq)
{
    const int b = blockIdx.x;
    const int t = threadIdx.x;
    __shared__ float red[256];

    float s = 0.f;
    for (int i = t; i < DD; i += 256) {
        float v = q[(size_t)b * DD + i];
        s += v * v;
        qbf[(size_t)b * DD + i] = (__bf16)v;
    }
    red[t] = s;
    __syncthreads();
    for (int off = 128; off > 0; off >>= 1) {
        if (t < off) red[t] += red[t + off];
        __syncthreads();
    }
    if (t == 0)
        invq[b] = 1.0f / fmaxf(sqrtf(red[0]), EPS);
}

// ---------------------------------------------------------------------------
// Kernel 2: keys ++ queue  ->  colbf[65792][768] bf16 + per-column inv norms.
// One wave per column; each lane converts 8 contiguous f32 -> b128 bf16 store.
// ---------------------------------------------------------------------------
__global__ __launch_bounds__(256) void prep_cols_kernel(
    const float* __restrict__ keys, const float* __restrict__ queue,
    __bf16* __restrict__ colbf, float* __restrict__ invc)
{
    const int wave = threadIdx.x >> 5;
    const int lane = threadIdx.x & 31;
    const int n = blockIdx.x * 8 + wave;           // n < NCOLS exactly

    const float* src = (n < BB) ? (keys + (size_t)n * DD)
                                : (queue + (size_t)(n - BB) * DD);
    __bf16* dst = colbf + (size_t)n * DD;

    float s = 0.f;
    #pragma unroll
    for (int i = 0; i < DD / 256; ++i) {           // 3 iterations
        const int idx = lane * 8 + i * 256;
        v4f a = *(const v4f*)(src + idx);
        v4f b = *(const v4f*)(src + idx + 4);
        s += a[0]*a[0] + a[1]*a[1] + a[2]*a[2] + a[3]*a[3];
        s += b[0]*b[0] + b[1]*b[1] + b[2]*b[2] + b[3]*b[3];
        v8bf o;
        o[0] = (__bf16)a[0]; o[1] = (__bf16)a[1];
        o[2] = (__bf16)a[2]; o[3] = (__bf16)a[3];
        o[4] = (__bf16)b[0]; o[5] = (__bf16)b[1];
        o[6] = (__bf16)b[2]; o[7] = (__bf16)b[3];
        *(v8bf*)(dst + idx) = o;
    }
    #pragma unroll
    for (int off = 16; off > 0; off >>= 1)
        s += __shfl_xor(s, off, 32);
    if (lane == 0)
        invc[n] = 1.0f / fmaxf(sqrtf(s), EPS);
}

// ---------------------------------------------------------------------------
// Fragment loaders (bf16 source).
// A (16x32): lane holds row M=l16; halves own K chunks {8h..8h+7, 16+8h..+7}.
// B (32x16): lane holds col N=l16; halves own K chunks {8h..8h+7, 16+8h..+7}.
// ---------------------------------------------------------------------------
__device__ __forceinline__ v16bf load_a(const __bf16* __restrict__ p)
{
    v8bf lo = *(const v8bf*)(p);
    v8bf hi = *(const v8bf*)(p + 16);
    return __builtin_shufflevector(lo, hi,
               0, 1, 2, 3, 4, 5, 6, 7, 8, 9, 10, 11, 12, 13, 14, 15);
}
__device__ __forceinline__ v16bf load_b(const __bf16* __restrict__ p)
{
    v8bf lo = *(const v8bf*)(p);
    v8bf hi = *(const v8bf*)(p + 16);
    return __builtin_shufflevector(lo, hi,
               0, 1, 2, 3, 4, 5, 6, 7, 8, 9, 10, 11, 12, 13, 14, 15);
}

#define WMMA_BF16(A, B, C) \
    __builtin_amdgcn_wmma_f32_16x16x32_bf16(false, (A), false, (B), (short)0, (C), false, false)

// ---------------------------------------------------------------------------
// Kernel 3: GEMM + fused online-softmax partials.
// Grid: 1028 blocks (one 64-col strip each, covers ALL 256 rows -> each
// column streams from HBM exactly once). Block: 8 waves; wave w owns rows
// [32w, 32w+32) as two 16-row fragments x four 16-col fragments
// -> 8 v_wmma_f32_16x16x32_bf16 per K-step per wave.
// Epilogue: per-row (max, sumexp) over the 64 cols via __shfl_xor (wave32),
// written to partial_m/partial_s [256][1028].
// ---------------------------------------------------------------------------
__global__ __launch_bounds__(256) void gemm_kernel(
    const __bf16* __restrict__ qbf, const __bf16* __restrict__ colbf,
    const float* __restrict__ invq, const float* __restrict__ invc,
    float* __restrict__ partial_m, float* __restrict__ partial_s)
{
    const int lane = threadIdx.x & 31;
    const int wave = threadIdx.x >> 5;
    const int half = lane >> 4;
    const int l16  = lane & 15;

    const int colBase = blockIdx.x * 64;
    const int rowBase = wave * 32;

    const __bf16* aRow0 = qbf + (size_t)(rowBase + l16) * DD + half * 8;
    const __bf16* aRow1 = aRow0 + (size_t)16 * DD;
    const __bf16* bCol0 = colbf + (size_t)(colBase + l16) * DD + half * 8;
    const __bf16* bCol1 = bCol0 + (size_t)16 * DD;
    const __bf16* bCol2 = bCol0 + (size_t)32 * DD;
    const __bf16* bCol3 = bCol0 + (size_t)48 * DD;

    v8f acc[2][4];
    #pragma unroll
    for (int t = 0; t < 2; ++t)
        #pragma unroll
        for (int c = 0; c < 4; ++c)
            acc[t][c] = (v8f){};

    #pragma unroll 2
    for (int k0 = 0; k0 < DD; k0 += 32) {
        v16bf a0 = load_a(aRow0 + k0);
        v16bf a1 = load_a(aRow1 + k0);
        v16bf b0 = load_b(bCol0 + k0);
        v16bf b1 = load_b(bCol1 + k0);
        v16bf b2 = load_b(bCol2 + k0);
        v16bf b3 = load_b(bCol3 + k0);

        acc[0][0] = WMMA_BF16(a0, b0, acc[0][0]);
        acc[1][0] = WMMA_BF16(a1, b0, acc[1][0]);
        acc[0][1] = WMMA_BF16(a0, b1, acc[0][1]);
        acc[1][1] = WMMA_BF16(a1, b1, acc[1][1]);
        acc[0][2] = WMMA_BF16(a0, b2, acc[0][2]);
        acc[1][2] = WMMA_BF16(a1, b2, acc[1][2]);
        acc[0][3] = WMMA_BF16(a0, b3, acc[0][3]);
        acc[1][3] = WMMA_BF16(a1, b3, acc[1][3]);
    }

    // Epilogue: scale to logits, then per-row online (m, s) over 64 cols.
    const float ic0 = invc[colBase +  0 + l16];
    const float ic1 = invc[colBase + 16 + l16];
    const float ic2 = invc[colBase + 32 + l16];
    const float ic3 = invc[colBase + 48 + l16];

    #pragma unroll
    for (int t = 0; t < 2; ++t) {
        #pragma unroll
        for (int v = 0; v < 8; ++v) {
            const int row = rowBase + t * 16 + v + 8 * half;
            const float scale = invq[row] * INV_TEMP;
            float x0 = acc[t][0][v] * scale * ic0;
            float x1 = acc[t][1][v] * scale * ic1;
            float x2 = acc[t][2][v] * scale * ic2;
            float x3 = acc[t][3][v] * scale * ic3;
            float m = fmaxf(fmaxf(x0, x1), fmaxf(x2, x3));
            float s = __expf(x0 - m) + __expf(x1 - m)
                    + __expf(x2 - m) + __expf(x3 - m);
            // reduce across the 16 lanes of this half (xor < 16 stays in half)
            #pragma unroll
            for (int off = 1; off <= 8; off <<= 1) {
                float m2 = __shfl_xor(m, off, 32);
                float s2 = __shfl_xor(s, off, 32);
                float M = fmaxf(m, m2);
                s = s * __expf(m - M) + s2 * __expf(m2 - M);
                m = M;
            }
            if (l16 == 0) {
                partial_m[(size_t)row * NBLK + blockIdx.x] = m;
                partial_s[(size_t)row * NBLK + blockIdx.x] = s;
            }
        }
    }
}

// ---------------------------------------------------------------------------
// Kernel 4: per-row merge of 1028 (m,s) partials -> lse; diagonal logit
// recomputed from the same bf16 data; rowloss[b] = logits[b,b] - lse.
// ---------------------------------------------------------------------------
__global__ __launch_bounds__(256) void reduce_kernel(
    const __bf16* __restrict__ qbf, const __bf16* __restrict__ colbf,
    const float* __restrict__ invq, const float* __restrict__ invc,
    const float* __restrict__ partial_m, const float* __restrict__ partial_s,
    float* __restrict__ rowloss)
{
    const int b = blockIdx.x;
    const int t = threadIdx.x;

    // diagonal: dot(qbf[b], colbf[b]) (col b < 256 is keys[b])
    float d = 0.f;
    for (int i = t; i < DD; i += 256)
        d += (float)qbf[(size_t)b * DD + i] * (float)colbf[(size_t)b * DD + i];
    __shared__ float red[256];
    red[t] = d;
    __syncthreads();
    for (int off = 128; off > 0; off >>= 1) {
        if (t < off) red[t] += red[t + off];
        __syncthreads();
    }
    const float diag = red[0] * invq[b] * invc[b] * INV_TEMP;

    // merge partials
    float m = -INFINITY, s = 0.f;
    for (int i = t; i < NBLK; i += 256) {
        float m2 = partial_m[(size_t)b * NBLK + i];
        float s2 = partial_s[(size_t)b * NBLK + i];
        float M = fmaxf(m, m2);
        s = s * __expf(m - M) + s2 * __expf(m2 - M);
        m = M;
    }
    __shared__ float sm[256], ss[256];
    sm[t] = m; ss[t] = s;
    __syncthreads();
    for (int off = 128; off > 0; off >>= 1) {
        if (t < off) {
            float m2 = sm[t + off], s2 = ss[t + off];
            float M = fmaxf(sm[t], m2);
            ss[t] = ss[t] * __expf(sm[t] - M) + s2 * __expf(m2 - M);
            sm[t] = M;
        }
        __syncthreads();
    }
    if (t == 0) {
        float lse = sm[0] + __logf(ss[0]);
        rowloss[b] = diag - lse;
    }
}

// ---------------------------------------------------------------------------
// Kernel 5: deterministic final mean (no float atomics -> bit-stable replays).
// ---------------------------------------------------------------------------
__global__ __launch_bounds__(256) void final_kernel(
    const float* __restrict__ rowloss, float* __restrict__ out)
{
    const int t = threadIdx.x;
    __shared__ float red[256];
    red[t] = rowloss[t];
    __syncthreads();
    for (int off = 128; off > 0; off >>= 1) {
        if (t < off) red[t] += red[t + off];
        __syncthreads();
    }
    if (t == 0)
        out[0] = -red[0] * (1.0f / (float)BB);
}

// ---------------------------------------------------------------------------
extern "C" void kernel_launch(void* const* d_in, const int* in_sizes, int n_in,
                              void* d_out, int out_size, void* d_ws, size_t ws_size,
                              hipStream_t stream)
{
    const float* query = (const float*)d_in[0];   // [256,768]
    const float* keys  = (const float*)d_in[1];   // [256,768]
    const float* queue = (const float*)d_in[2];   // [64,1024,768]
    float* out = (float*)d_out;

    char* ws = (char*)d_ws;
    size_t off = 0;
    __bf16* colbf = (__bf16*)(ws + off); off += (size_t)NCOLS * DD * sizeof(__bf16);
    __bf16* qbf   = (__bf16*)(ws + off); off += (size_t)BB * DD * sizeof(__bf16);
    float* invq   = (float*)(ws + off);  off += (size_t)BB * sizeof(float);
    float* invc   = (float*)(ws + off);  off += (size_t)NCOLS * sizeof(float);
    float* part_m = (float*)(ws + off);  off += (size_t)BB * NBLK * sizeof(float);
    float* part_s = (float*)(ws + off);  off += (size_t)BB * NBLK * sizeof(float);
    float* rowls  = (float*)(ws + off);

    prep_q_kernel<<<BB, 256, 0, stream>>>(query, qbf, invq);
    prep_cols_kernel<<<NCOLS / 8, 256, 0, stream>>>(keys, queue, colbf, invc);
    gemm_kernel<<<NBLK, 256, 0, stream>>>(qbf, colbf, invq, invc, part_m, part_s);
    reduce_kernel<<<BB, 256, 0, stream>>>(qbf, colbf, invq, invc,
                                          part_m, part_s, rowls);
    final_kernel<<<1, 256, 0, stream>>>(rowls, out);
}
